// Block_44951127720380
// MI455X (gfx1250) — compile-verified, matrix operand order checked by reading, not run on hardware
//
#include <hip/hip_runtime.h>
#include <hip/hip_bf16.h>
#include <math.h>

// ---------------- problem constants ----------------
#define BB 2
#define TT 2048
#define CC 1024
#define HH 16
#define DD 64
#define RR (BB*TT)   // 4096 rows

typedef __attribute__((ext_vector_type(16))) _Float16 v16h;
typedef __attribute__((ext_vector_type(8)))  float    v8f;

// ---------------- WMMA helpers ----------------
__device__ __forceinline__ v8f wmma16x16x32(v16h a, v16h b, v8f c) {
  // (neg_a, A, neg_b, B, c_mod, C, reuse_a, reuse_b)
  return __builtin_amdgcn_wmma_f32_16x16x32_f16(false, a, false, b, (short)0, c,
                                                false, false);
}

// Fragment (16 rows x 32 k, f16) from a row-major matrix [rows, ld] at (row0, k0).
// ISA 16-bit A layout: lanes 0-15 -> M=lane, K in {0..7,16..23}; lanes 16-31 ->
// M=lane-16, K in {8..15,24..31}.  Contiguous-in-k => two 16B loads per lane.
// Used for A fragments (rows = M) and, on a transposed matrix, for B fragments
// (rows = N) since B[k][n] with lane = n has the same per-lane k pattern.
__device__ __forceinline__ v16h frag_MK(const _Float16* __restrict__ p, int ld,
                                        int row0, int k0, int lane) {
  const int m  = row0 + (lane & 15);
  const int kb = k0 + ((lane >> 4) << 3);
  v16h r = {};
#pragma unroll
  for (int e = 0; e < 16; ++e) {
    const int k = kb + (e & 7) + ((e >> 3) << 4);
    r[e] = p[(size_t)m * ld + k];
  }
  return r;
}

__device__ __forceinline__ float gelu_exact(float x) {
  return 0.5f * x * (1.0f + erff(x * 0.70710678118654752440f));
}

// block-wide sum over 256 threads
__device__ __forceinline__ float block_sum256(float v) {
  __shared__ float sm[256];
  const int tid = threadIdx.x;
  sm[tid] = v;
  __syncthreads();
#pragma unroll
  for (int s = 128; s > 0; s >>= 1) {
    if (tid < s) sm[tid] += sm[tid + s];
    __syncthreads();
  }
  float r = sm[0];
  __syncthreads();
  return r;
}

// ---------------- weight conversion: f32 [K,N] -> f16 transposed [N,K] ----------------
__global__ void k_cvt_w_t(const float* __restrict__ src, _Float16* __restrict__ dst,
                          int K, int N) {
  const size_t i = (size_t)blockIdx.x * 256 + threadIdx.x;  // over K*N
  if (i >= (size_t)K * N) return;
  const int k = (int)(i / N);
  const int n = (int)(i % N);
  dst[(size_t)n * K + k] = (_Float16)src[i];
}

// ---------------- LayerNorm (block per row, 256 threads) ----------------
__global__ void k_layernorm(const float* __restrict__ x, const float* __restrict__ w,
                            const float* __restrict__ bv, float* __restrict__ o32,
                            _Float16* __restrict__ o16) {
  const int row = blockIdx.x;
  const int tid = threadIdx.x;
  const float* xr = x + (size_t)row * CC;
  float v[4];
  float s = 0.f;
#pragma unroll
  for (int j = 0; j < 4; ++j) { v[j] = xr[tid + 256 * j]; s += v[j]; }
  const float mu = block_sum256(s) * (1.0f / CC);
  float sq = 0.f;
#pragma unroll
  for (int j = 0; j < 4; ++j) { const float d = v[j] - mu; sq += d * d; }
  const float var = block_sum256(sq) * (1.0f / CC);
  const float inv = rsqrtf(var + 1e-5f);
#pragma unroll
  for (int j = 0; j < 4; ++j) {
    const int c = tid + 256 * j;
    const float o = (v[j] - mu) * inv * w[c] + bv[c];
    if (o32) o32[(size_t)row * CC + c] = o;
    if (o16) o16[(size_t)row * CC + c] = (_Float16)o;
  }
}

// ---------------- logmap(ref, xn) with ref = shift(xn) ----------------
__global__ void k_logmap(const float* __restrict__ xn, _Float16* __restrict__ xt) {
  const int row = blockIdx.x;
  const int t   = row % TT;
  const int tid = threadIdx.x;
  const float* xr = xn + (size_t)row * CC;
  const float* rr = xn + (size_t)(row - 1) * CC;  // only read when t > 0
  float xv[4], rv[4];
  float srn = 0.f, syn = 0.f, sip = 0.f;
#pragma unroll
  for (int j = 0; j < 4; ++j) {
    const int c = tid + 256 * j;
    xv[j] = xr[c];
    rv[j] = (t > 0) ? rr[c] : 0.0f;
    srn += rv[j] * rv[j];
    syn += xv[j] * xv[j];
    sip += -rv[j] * xv[j];        // ip of (-ref, xn)
  }
  const float rn2 = block_sum256(srn);   // |ref|^2 = |-ref|^2
  const float yn2 = block_sum256(syn);   // |xn|^2
  const float ip  = block_sum256(sip);
  // mobius_addition(-ref, xn), c = 1
  const float A   = 1.0f + 2.0f * ip + yn2;
  const float Bc  = 1.0f - rn2;
  const float den = 1.0f + 2.0f * ip + rn2 * yn2;
  float mv[4];
  float smn = 0.f;
#pragma unroll
  for (int j = 0; j < 4; ++j) {
    mv[j] = (A * (-rv[j]) + Bc * xv[j]) / den;
    smn += mv[j] * mv[j];
  }
  const float mn  = sqrtf(block_sum256(smn));
  const float lam = 2.0f / (1.0f + rn2);
  const float cf  = 2.0f / lam;
  float arg = sqrtf(mn);                 // sqrt(c*mn), c=1
  arg = fminf(arg, 0.999f);
  const float coef = cf * atanhf(arg) / mn;
#pragma unroll
  for (int j = 0; j < 4; ++j) {
    const int c = tid + 256 * j;
    xt[(size_t)row * CC + c] = (_Float16)(coef * mv[j]);
  }
}

// ---------------- expmap(ref, y) with ref = shift(xn) ----------------
__global__ void k_expmap(const float* __restrict__ xn, const float* __restrict__ y,
                         _Float16* __restrict__ z) {
  const int row = blockIdx.x;
  const int t   = row % TT;
  const int tid = threadIdx.x;
  const float* yr = y + (size_t)row * CC;
  const float* rr = xn + (size_t)(row - 1) * CC;
  float vv[4], rv[4];
  float srn = 0.f, svn = 0.f, sip = 0.f;
#pragma unroll
  for (int j = 0; j < 4; ++j) {
    const int c = tid + 256 * j;
    vv[j] = yr[c];
    rv[j] = (t > 0) ? rr[c] : 0.0f;
    srn += rv[j] * rv[j];
    svn += vv[j] * vv[j];
    sip += rv[j] * vv[j];
  }
  const float rn2 = block_sum256(srn);
  const float vn2 = block_sum256(svn);
  const float ipv = block_sum256(sip);
  const float lam = 2.0f / (1.0f + rn2);
  const float vn  = sqrtf(vn2);
  const float sfac = tanhf(sqrtf(lam * vn2 * 0.5f)) / vn;  // c = 1
  const float yn2 = sfac * sfac * vn2;  // |second|^2
  const float ip  = sfac * ipv;         // <ref, second>
  const float A   = 1.0f + 2.0f * ip + yn2;
  const float Bc  = 1.0f - rn2;
  const float den = 1.0f + 2.0f * ip + rn2 * yn2;
#pragma unroll
  for (int j = 0; j < 4; ++j) {
    const int c = tid + 256 * j;
    const float o = (A * rv[j] + Bc * sfac * vv[j]) / den;
    z[(size_t)row * CC + c] = (_Float16)o;
  }
}

// ------- repack qkv -> q,k [B,H,T,D] f16 (q * 1/sqrt(D)), v transposed [B,H,D,T] -------
__global__ void k_repack_qkv(const float* __restrict__ qkv, _Float16* __restrict__ q16,
                             _Float16* __restrict__ k16, _Float16* __restrict__ vt16) {
  const size_t i = (size_t)blockIdx.x * 256 + threadIdx.x;  // over B*T*C
  const int d = (int)(i % DD);
  const int h = (int)((i / DD) % HH);
  const int t = (int)((i / CC) % TT);
  const int b = (int)(i / ((size_t)CC * TT));
  const int bh = b * HH + h;
  const size_t src = ((size_t)b * TT + t) * (3 * CC) + h * DD + d;
  const size_t dst = ((size_t)bh * TT + t) * DD + d;
  q16[dst] = (_Float16)(qkv[src] * 0.125f);      // 1/sqrt(64)
  k16[dst] = (_Float16)qkv[src + CC];
  vt16[((size_t)bh * DD + d) * TT + t] = (_Float16)qkv[src + 2 * CC];
}

// ------- WMMA GEMM: out = A(f16)[M,K] @ W (given as WT f16 [N,K]) + bias, epilogue -------
// Block: 256 threads = 8 waves as 2(M) x 4(N); tile 32M x 256N; wave strip 16M x 64N.
// mode 0: out32 = v + bias
// mode 1: out32 = v + bias + resid
// mode 2: out16 = gelu(v + bias)
__global__ void k_gemm(const _Float16* __restrict__ A, const _Float16* __restrict__ WT,
                       const float* __restrict__ bias, const float* __restrict__ resid,
                       float* __restrict__ out32, _Float16* __restrict__ out16,
                       int M, int N, int K, int mode) {
  const int lane = threadIdx.x & 31;
  const int wave = threadIdx.x >> 5;
  const int m0 = blockIdx.y * 32 + (wave >> 2) * 16;
  const int n0 = blockIdx.x * 256 + (wave & 3) * 64;
  v8f acc[4] = {v8f{}, v8f{}, v8f{}, v8f{}};
  for (int k0 = 0; k0 < K; k0 += 32) {
    if (k0 + 32 < K) {  // prefetch next K-step (global_prefetch_b8)
      __builtin_prefetch(&A[(size_t)(m0 + (lane & 15)) * K + k0 + 32], 0, 1);
      __builtin_prefetch(&WT[(size_t)(n0 + (lane & 15)) * K + k0 + 32], 0, 1);
    }
    const v16h a = frag_MK(A, K, m0, k0, lane);
#pragma unroll
    for (int j = 0; j < 4; ++j) {
      const v16h b = frag_MK(WT, K, n0 + 16 * j, k0, lane);  // B[k][n] from W^T
      acc[j] = wmma16x16x32(a, b, acc[j]);
    }
  }
  const int nl = lane & 15;
  const int rb = m0 + ((lane >> 4) << 3);
#pragma unroll
  for (int j = 0; j < 4; ++j) {
    const int n = n0 + 16 * j + nl;
    const float bs = bias[n];
#pragma unroll
    for (int i = 0; i < 8; ++i) {
      float v = acc[j][i] + bs;
      const size_t idx = (size_t)(rb + i) * N + n;
      if (mode == 2) {
        out16[idx] = (_Float16)gelu_exact(v);
      } else {
        if (mode == 1) v += resid[idx];
        out32[idx] = v;
      }
    }
  }
}

// ---------------- causal flash attention, one wave per 16-query tile ----------------
__global__ void k_attn(const _Float16* __restrict__ q16, const _Float16* __restrict__ k16,
                       const _Float16* __restrict__ vt16, float* __restrict__ ybuf) {
  const int lane = threadIdx.x & 31;
  const int q0   = blockIdx.x * 16;       // query tile
  const int bh   = blockIdx.y;            // b*H + h
  const int b    = bh / HH;
  const int h    = bh % HH;
  const _Float16* qb  = q16  + (size_t)bh * TT * DD;
  const _Float16* kb  = k16  + (size_t)bh * TT * DD;
  const _Float16* vtb = vt16 + (size_t)bh * DD * TT;   // [D, T]

  const v16h aq0 = frag_MK(qb, DD, q0, 0, lane);
  const v16h aq1 = frag_MK(qb, DD, q0, 32, lane);

  v8f acc[4] = {v8f{}, v8f{}, v8f{}, v8f{}};   // 16 x 64 output (4 dim groups)
  float mrow[8], lrow[8];
#pragma unroll
  for (int i = 0; i < 8; ++i) { mrow[i] = -1e30f; lrow[i] = 0.f; }

  __shared__ _Float16 lds_p[16 * 32];          // probs tile, one wave per block

  const int colk  = lane & 15;
  const int rbase = (lane >> 4) << 3;
  const int nsteps = (q0 + 15) / 32 + 1;

  for (int s = 0; s < nsteps; ++s) {
    const int kk = s * 32;
    // scores S[16q x 32k] as two 16x16 tiles
    v8f s0 = {}, s1 = {};
    {
      const v16h b0 = frag_MK(kb, DD, kk,      0,  lane);  // K^T fragment
      const v16h b1 = frag_MK(kb, DD, kk,      32, lane);
      s0 = wmma16x16x32(aq0, b0, s0);
      s0 = wmma16x16x32(aq1, b1, s0);
      const v16h b2 = frag_MK(kb, DD, kk + 16, 0,  lane);
      const v16h b3 = frag_MK(kb, DD, kk + 16, 32, lane);
      s1 = wmma16x16x32(aq0, b2, s1);
      s1 = wmma16x16x32(aq1, b3, s1);
    }
    // causal mask + online softmax per row
#pragma unroll
    for (int i = 0; i < 8; ++i) {
      const int qrow = q0 + rbase + i;
      if (kk + colk      > qrow) s0[i] = -1e30f;
      if (kk + 16 + colk > qrow) s1[i] = -1e30f;
      float mx = fmaxf(s0[i], s1[i]);
#pragma unroll
      for (int off = 1; off < 16; off <<= 1)
        mx = fmaxf(mx, __shfl_xor(mx, off, 32));
      const float mnew  = fmaxf(mrow[i], mx);
      const float p0    = __expf(s0[i] - mnew);
      const float p1    = __expf(s1[i] - mnew);
      const float alpha = __expf(mrow[i] - mnew);
      float psum = p0 + p1;
#pragma unroll
      for (int off = 1; off < 16; off <<= 1)
        psum += __shfl_xor(psum, off, 32);
      lrow[i] = lrow[i] * alpha + psum;
      mrow[i] = mnew;
#pragma unroll
      for (int f = 0; f < 4; ++f) acc[f][i] *= alpha;
      const int r = rbase + i;
      lds_p[r * 32 + colk]      = (_Float16)p0;
      lds_p[r * 32 + 16 + colk] = (_Float16)p1;
    }
    // wait for this wave's LDS stores before fragment reload (single wave/block)
    asm volatile("s_wait_dscnt 0" ::: "memory");
    const v16h ap = frag_MK((const _Float16*)lds_p, 32, 0, 0, lane);
#pragma unroll
    for (int f = 0; f < 4; ++f) {
      // V^T fragment: lane = dim (f*16 + nl), contiguous keys kk..kk+31
      const v16h bv = frag_MK(vtb, TT, f * 16, kk, lane);
      acc[f] = wmma16x16x32(ap, bv, acc[f]);
    }
    asm volatile("s_wait_dscnt 0" ::: "memory");
  }

  // write y[b, t, h*D + d]
#pragma unroll
  for (int f = 0; f < 4; ++f)
#pragma unroll
    for (int i = 0; i < 8; ++i) {
      const int trow = q0 + rbase + i;
      const float val = acc[f][i] / lrow[i];
      ybuf[((size_t)b * TT + trow) * CC + h * DD + f * 16 + colk] = val;
    }
}

// ---------------- host orchestration ----------------
extern "C" void kernel_launch(void* const* d_in, const int* in_sizes, int n_in,
                              void* d_out, int out_size, void* d_ws, size_t ws_size,
                              hipStream_t stream) {
  const float* x      = (const float*)d_in[0];
  const float* ln1w   = (const float*)d_in[1];
  const float* ln1b   = (const float*)d_in[2];
  const float* Wattn  = (const float*)d_in[3];
  const float* battn  = (const float*)d_in[4];
  const float* Waproj = (const float*)d_in[5];
  const float* baproj = (const float*)d_in[6];
  const float* ln2w   = (const float*)d_in[7];
  const float* ln2b   = (const float*)d_in[8];
  const float* Wfc    = (const float*)d_in[9];
  const float* bfc    = (const float*)d_in[10];
  const float* Wmproj = (const float*)d_in[11];
  const float* bmproj = (const float*)d_in[12];
  float* out = (float*)d_out;

  char* p = (char*)d_ws;
  auto alloc = [&](size_t bytes) -> char* {
    char* r = p;
    p += (bytes + 255) & ~(size_t)255;
    return r;
  };
  float*    xn    = (float*)   alloc((size_t)RR * CC * 4);
  _Float16* xt16  = (_Float16*)alloc((size_t)RR * CC * 2);
  _Float16* wat16 = (_Float16*)alloc((size_t)CC * 3 * CC * 2);  // [3C, C] = Wattn^T
  _Float16* wap16 = (_Float16*)alloc((size_t)CC * CC * 2);      // [C, C]  = Waproj^T
  _Float16* wfc16 = (_Float16*)alloc((size_t)CC * 4 * CC * 2);  // [4C, C] = Wfc^T
  _Float16* wmp16 = (_Float16*)alloc((size_t)4 * CC * CC * 2);  // [C, 4C] = Wmproj^T
  float*    qkv32 = (float*)   alloc((size_t)RR * 3 * CC * 4);
  _Float16* q16   = (_Float16*)alloc((size_t)RR * CC * 2);
  _Float16* k16   = (_Float16*)alloc((size_t)RR * CC * 2);
  _Float16* vt16  = (_Float16*)alloc((size_t)RR * CC * 2);
  float*    ybuf  = (float*)   alloc((size_t)RR * CC * 4);
  _Float16* z16   = (_Float16*)alloc((size_t)RR * CC * 2);
  float*    r1    = (float*)   alloc((size_t)RR * CC * 4);
  _Float16* h16   = (_Float16*)alloc((size_t)RR * CC * 2);
  _Float16* g16   = (_Float16*)alloc((size_t)RR * 4 * CC * 2);

  auto cvt_t = [&](const float* s, _Float16* d, int K, int N) {
    const size_t n = (size_t)K * N;
    k_cvt_w_t<<<(unsigned)((n + 255) / 256), 256, 0, stream>>>(s, d, K, N);
  };
  cvt_t(Wattn,  wat16, CC, 3 * CC);
  cvt_t(Waproj, wap16, CC, CC);
  cvt_t(Wfc,    wfc16, CC, 4 * CC);
  cvt_t(Wmproj, wmp16, 4 * CC, CC);

  // ---- attention branch ----
  k_layernorm<<<RR, 256, 0, stream>>>(x, ln1w, ln1b, xn, ((_Float16*)nullptr));
  k_logmap<<<RR, 256, 0, stream>>>(xn, xt16);
  k_gemm<<<dim3(3 * CC / 256, RR / 32), 256, 0, stream>>>(
      xt16, wat16, battn, nullptr, qkv32, nullptr, RR, 3 * CC, CC, 0);
  k_repack_qkv<<<(unsigned)((size_t)RR * CC / 256), 256, 0, stream>>>(qkv32, q16, k16, vt16);
  k_attn<<<dim3(TT / 16, BB * HH), 32, 0, stream>>>(q16, k16, vt16, ybuf);
  k_expmap<<<RR, 256, 0, stream>>>(xn, ybuf, z16);
  k_gemm<<<dim3(CC / 256, RR / 32), 256, 0, stream>>>(
      z16, wap16, baproj, x, r1, nullptr, RR, CC, CC, 1);

  // ---- MLP branch ----
  k_layernorm<<<RR, 256, 0, stream>>>(r1, ln2w, ln2b, (float*)nullptr, h16);
  k_gemm<<<dim3(4 * CC / 256, RR / 32), 256, 0, stream>>>(
      h16, wfc16, bfc, nullptr, nullptr, g16, RR, 4 * CC, CC, 2);
  k_gemm<<<dim3(CC / 256, RR / 32), 256, 0, stream>>>(
      g16, wmp16, bmproj, r1, out, nullptr, RR, CC, 4 * CC, 1);
}
